// CustomGAT_50302656971211
// MI455X (gfx1250) — compile-verified, matrix operand order checked by reading, not run on hardware
//
#include <hip/hip_runtime.h>
#include <hip/hip_bf16.h>
#include <math.h>

// ---------------------------------------------------------------------------
// Types for CDNA5 WMMA (wave32): v16bf A/B fragments, v8f accumulator.
// ---------------------------------------------------------------------------
typedef __attribute__((ext_vector_type(16))) __bf16 v16bf;
typedef __attribute__((ext_vector_type(8)))  float  v8f;
typedef __attribute__((ext_vector_type(4)))  int    v4i;

struct alignas(16) U4 { unsigned int x, y, z, w; };   // trivial 16B chunk
union Frag16 { v16bf v; U4 q[2]; };

#define GN_NODES 64
#define ROWS 1024            // B*N
#define FEAT 4096
#define HEADS 4
#define OUTH 1024

// GEMM tiling: C[1024,4096] = A_bf16[1024,4096] x Wt_bf16(n-major)[4096,4096]
#define TM 128
#define TN 128
#define TK 32

// gfx1250 async global->LDS path (ASYNCcnt), guarded so absence falls back.
#if defined(__HIP_DEVICE_COMPILE__) && \
    __has_builtin(__builtin_amdgcn_global_load_async_to_lds_b128) && \
    __has_builtin(__builtin_amdgcn_s_wait_asynccnt)
#define USE_ASYNC_LDS 1
typedef __attribute__((address_space(1))) v4i gv4i;   // global 16B payload
typedef __attribute__((address_space(3))) v4i lv4i;   // LDS    16B payload
__device__ __forceinline__ void async_cp32(const __bf16* g, __bf16* l) {
  // 32 bytes = two b128 async copies (GLOBAL_LOAD_ASYNC_TO_LDS_B128)
  __builtin_amdgcn_global_load_async_to_lds_b128((gv4i*)g,       (lv4i*)l,       0, 0);
  __builtin_amdgcn_global_load_async_to_lds_b128((gv4i*)(g + 8), (lv4i*)(l + 8), 0, 0);
}
#else
#define USE_ASYNC_LDS 0
#endif

// ---------------------------------------------------------------------------
// Kernel 0: W f32 [k][n] -> Wt bf16 [n][k]  (tiled 32x32 transpose+convert)
// ---------------------------------------------------------------------------
__global__ __launch_bounds__(256) void wconv_k(
    const float* __restrict__ W, unsigned short* __restrict__ Wt_us)
{
  __bf16* Wt = reinterpret_cast<__bf16*>(Wt_us);
  __shared__ float sx[32][33];
  const int tx = threadIdx.x & 31, ty = threadIdx.x >> 5;   // 32 x 8
  const int kT = (blockIdx.x & 127) * 32;
  const int nT = (blockIdx.x >> 7) * 32;
#pragma unroll
  for (int q = 0; q < 4; ++q)
    sx[ty + 8 * q][tx] = W[(size_t)(kT + ty + 8 * q) * FEAT + nT + tx];
  __syncthreads();
#pragma unroll
  for (int q = 0; q < 4; ++q)
    Wt[(size_t)(nT + ty + 8 * q) * FEAT + kT + tx] = (__bf16)sx[tx][ty + 8 * q];
}

// ---------------------------------------------------------------------------
// Kernel 1: conv1  [1024,3,16,16] -> [1024,8,16,16], 3x3 SAME, ReLU
// ---------------------------------------------------------------------------
__global__ __launch_bounds__(256) void conv1_k(
    const float* __restrict__ x, const float* __restrict__ w,
    const float* __restrict__ b, float* __restrict__ out)
{
  const int m  = blockIdx.x;
  const int px = threadIdx.x;
  const int hh = px >> 4, ww = px & 15;

  __shared__ float sx[3][256];
  __shared__ float sw[8 * 3 * 9];
  __shared__ float sb[8];
  for (int i = px; i < 768; i += 256) sx[i >> 8][i & 255] = x[(size_t)m * 768 + i];
  if (px < 216) sw[px] = w[px];
  if (px < 8)   sb[px] = b[px];
  __syncthreads();

  for (int co = 0; co < 8; ++co) {
    float s = sb[co];
#pragma unroll
    for (int ci = 0; ci < 3; ++ci)
#pragma unroll
      for (int kh = -1; kh <= 1; ++kh)
#pragma unroll
        for (int kw = -1; kw <= 1; ++kw) {
          int y = hh + kh, xx = ww + kw;
          if (y >= 0 && y < 16 && xx >= 0 && xx < 16)
            s += sx[ci][y * 16 + xx] * sw[((co * 3 + ci) * 3 + kh + 1) * 3 + kw + 1];
        }
    out[(size_t)m * 2048 + co * 256 + px] = fmaxf(s, 0.f);
  }
}

// ---------------------------------------------------------------------------
// Kernel 2: conv2  [1024,8,16,16] -> bf16 [1024,4096]
// ---------------------------------------------------------------------------
__global__ __launch_bounds__(256) void conv2_k(
    const float* __restrict__ in, const float* __restrict__ w,
    const float* __restrict__ b, unsigned short* __restrict__ out_us)
{
  __bf16* out = reinterpret_cast<__bf16*>(out_us);
  const int m  = blockIdx.x;
  const int px = threadIdx.x;
  const int hh = px >> 4, ww = px & 15;

  __shared__ float sx[8][256];
  __shared__ float sw[16 * 8 * 9];
  __shared__ float sb[16];
  for (int i = px; i < 2048; i += 256) sx[i >> 8][i & 255] = in[(size_t)m * 2048 + i];
  for (int i = px; i < 1152; i += 256) sw[i] = w[i];
  if (px < 16) sb[px] = b[px];
  __syncthreads();

  for (int co = 0; co < 16; ++co) {
    float s = sb[co];
#pragma unroll
    for (int ci = 0; ci < 8; ++ci)
#pragma unroll
      for (int kh = -1; kh <= 1; ++kh)
#pragma unroll
        for (int kw = -1; kw <= 1; ++kw) {
          int y = hh + kh, xx = ww + kw;
          if (y >= 0 && y < 16 && xx >= 0 && xx < 16)
            s += sx[ci][y * 16 + xx] * sw[((co * 8 + ci) * 3 + kh + 1) * 3 + kw + 1];
        }
    out[(size_t)m * 4096 + co * 256 + px] = (__bf16)fmaxf(s, 0.f);
  }
}

// ---------------------------------------------------------------------------
// Kernel 3: WMMA bf16 GEMM  C[1024,4096] = A[1024,4096] x Wt[4096,4096]^T
// A bf16 row-major; Wt bf16 N-major. Double-buffered LDS, 1 barrier/K-step.
// Block = 256 threads = 8 wave32; block tile 128x128; wave tile 32x64.
// ---------------------------------------------------------------------------
__global__ __launch_bounds__(256) void gemm_bf16_wmma(
    const unsigned short* __restrict__ A_us,
    const unsigned short* __restrict__ Bt_us,
    float* __restrict__ C)
{
  const __bf16* A  = reinterpret_cast<const __bf16*>(A_us);
  const __bf16* Bt = reinterpret_cast<const __bf16*>(Bt_us);

  __shared__ __bf16 As[2][TM][TK];   // 2 x 8 KB
  __shared__ __bf16 Bs[2][TN][TK];   // 2 x 8 KB

  const int tid    = threadIdx.x;
  const int lane   = tid & 31;
  const int wave   = tid >> 5;
  const int mBlock = (blockIdx.x & 7) * TM;        // 1024/128 = 8 M-tiles
  const int nBlock = (blockIdx.x >> 3) * TN;       // 4096/128 = 32 N-tiles
  const int mw     = (wave & 3) * 32;              // wave M offset
  const int nw     = (wave >> 2) * 64;             // wave N offset
  const int lrow   = lane & 15;
  const int khalf  = (lane >> 4) * 8;              // ISA 16-bit A/B layout

  v8f acc[2][4];
#pragma unroll
  for (int i = 0; i < 2; ++i)
#pragma unroll
    for (int j = 0; j < 4; ++j) acc[i][j] = v8f{};

  // Staging: 128 rows x 32 bf16; 2 threads/row, 16 bf16 (32 B) each.
  const int row128 = tid >> 1;
  const int half   = (tid & 1) * 16;
  const __bf16* aSrc = A  + (size_t)(mBlock + row128) * FEAT + half;
  const __bf16* bSrc = Bt + (size_t)(nBlock + row128) * FEAT + half;

#if USE_ASYNC_LDS
  // ---- prologue: async-stage K-tile 0 into buffer 0 ----
  async_cp32(aSrc, &As[0][row128][half]);
  async_cp32(bSrc, &Bs[0][row128][half]);
  __builtin_amdgcn_s_wait_asynccnt(0);
  __syncthreads();

  for (int kb = 0; kb < FEAT; kb += TK) {
    const int  buf = (kb >> 5) & 1;
    const bool nxt = (kb + TK) < FEAT;
    if (nxt) {   // fill the other buffer while this one is consumed
      async_cp32(aSrc + kb + TK, &As[buf ^ 1][row128][half]);
      async_cp32(bSrc + kb + TK, &Bs[buf ^ 1][row128][half]);
    }

    Frag16 aF[2], bF[4];
#pragma unroll
    for (int i = 0; i < 2; ++i) {
      const int r = mw + i * 16 + lrow;
      aF[i].q[0] = *(const U4*)&As[buf][r][khalf];
      aF[i].q[1] = *(const U4*)&As[buf][r][khalf + 16];
    }
#pragma unroll
    for (int j = 0; j < 4; ++j) {
      const int c = nw + j * 16 + lrow;
      bF[j].q[0] = *(const U4*)&Bs[buf][c][khalf];
      bF[j].q[1] = *(const U4*)&Bs[buf][c][khalf + 16];
    }
#pragma unroll
    for (int i = 0; i < 2; ++i)
#pragma unroll
      for (int j = 0; j < 4; ++j)
        acc[i][j] = __builtin_amdgcn_wmma_f32_16x16x32_bf16(
            false, aF[i].v, false, bF[j].v, (short)0, acc[i][j], false, false);

    if (nxt) __builtin_amdgcn_s_wait_asynccnt(0);
    __syncthreads();
  }
#else
  // ---- fallback: register-staged double buffer ----
  U4 a0 = ((const U4*)aSrc)[0], a1 = ((const U4*)aSrc)[1];
  U4 b0 = ((const U4*)bSrc)[0], b1 = ((const U4*)bSrc)[1];
  *(U4*)&As[0][row128][half]     = a0;
  *(U4*)&As[0][row128][half + 8] = a1;
  *(U4*)&Bs[0][row128][half]     = b0;
  *(U4*)&Bs[0][row128][half + 8] = b1;
  __syncthreads();

  for (int kb = 0; kb < FEAT; kb += TK) {
    const int  buf = (kb >> 5) & 1;
    const bool nxt = (kb + TK) < FEAT;
    if (nxt) {
      const __bf16* an = aSrc + kb + TK;
      const __bf16* bn = bSrc + kb + TK;
      a0 = ((const U4*)an)[0];  a1 = ((const U4*)an)[1];
      b0 = ((const U4*)bn)[0];  b1 = ((const U4*)bn)[1];
      if (kb + 2 * TK < FEAT) {   // gfx1250 global_prefetch_b8
        __builtin_prefetch(aSrc + kb + 2 * TK, 0, 1);
        __builtin_prefetch(bSrc + kb + 2 * TK, 0, 1);
      }
    }

    Frag16 aF[2], bF[4];
#pragma unroll
    for (int i = 0; i < 2; ++i) {
      const int r = mw + i * 16 + lrow;
      aF[i].q[0] = *(const U4*)&As[buf][r][khalf];
      aF[i].q[1] = *(const U4*)&As[buf][r][khalf + 16];
    }
#pragma unroll
    for (int j = 0; j < 4; ++j) {
      const int c = nw + j * 16 + lrow;
      bF[j].q[0] = *(const U4*)&Bs[buf][c][khalf];
      bF[j].q[1] = *(const U4*)&Bs[buf][c][khalf + 16];
    }
#pragma unroll
    for (int i = 0; i < 2; ++i)
#pragma unroll
      for (int j = 0; j < 4; ++j)
        acc[i][j] = __builtin_amdgcn_wmma_f32_16x16x32_bf16(
            false, aF[i].v, false, bF[j].v, (short)0, acc[i][j], false, false);

    if (nxt) {
      *(U4*)&As[buf ^ 1][row128][half]     = a0;
      *(U4*)&As[buf ^ 1][row128][half + 8] = a1;
      *(U4*)&Bs[buf ^ 1][row128][half]     = b0;
      *(U4*)&Bs[buf ^ 1][row128][half + 8] = b1;
    }
    __syncthreads();
  }
#endif

  // ---- store C: VGPR v -> row v (lanes 0-15) / 8+v (lanes 16-31) ----
  const int rOff = (lane >> 4) * 8;
#pragma unroll
  for (int i = 0; i < 2; ++i) {
#pragma unroll
    for (int j = 0; j < 4; ++j) {
      const int col     = nBlock + nw + j * 16 + lrow;
      const int rowBase = mBlock + mw + i * 16 + rOff;
#pragma unroll
      for (int v = 0; v < 8; ++v)
        C[(size_t)(rowBase + v) * FEAT + col] = acc[i][j][v];
    }
  }
}

// ---------------------------------------------------------------------------
// Kernel 4: attention coefficients  a_s/a_d [1024][4]
// ---------------------------------------------------------------------------
__global__ __launch_bounds__(256) void attn_coef_k(
    const float* __restrict__ xl, const float* __restrict__ asrc,
    const float* __restrict__ adst, float* __restrict__ as_o,
    float* __restrict__ ad_o)
{
  const int r = blockIdx.x, t = threadIdx.x;
  __shared__ float red[256];
  for (int h = 0; h < HEADS; ++h) {
    float s = 0.f, d = 0.f;
#pragma unroll
    for (int i = 0; i < 4; ++i) {
      const int o = t + 256 * i;
      const float v = xl[(size_t)r * FEAT + h * OUTH + o];
      s += v * asrc[h * OUTH + o];
      d += v * adst[h * OUTH + o];
    }
    red[t] = s; __syncthreads();
    for (int st = 128; st > 0; st >>= 1) { if (t < st) red[t] += red[t + st]; __syncthreads(); }
    if (t == 0) as_o[r * 4 + h] = red[0];
    __syncthreads();
    red[t] = d; __syncthreads();
    for (int st = 128; st > 0; st >>= 1) { if (t < st) red[t] += red[t + st]; __syncthreads(); }
    if (t == 0) ad_o[r * 4 + h] = red[0];
    __syncthreads();
  }
}

// ---------------------------------------------------------------------------
// Kernel 5: masked softmax + aggregate + bias + ELU.  Block per (b,i,h).
// ---------------------------------------------------------------------------
__global__ __launch_bounds__(256) void attn_agg_k(
    const float* __restrict__ xl, const float* __restrict__ pos,
    const float* __restrict__ as_i, const float* __restrict__ ad_i,
    const float* __restrict__ bias, float* __restrict__ yf,
    unsigned short* __restrict__ ybf_us)
{
  __bf16* ybf = reinterpret_cast<__bf16*>(ybf_us);
  const int blk = blockIdx.x;
  const int b = blk >> 8, i = (blk >> 2) & 63, h = blk & 3;
  const int t = threadIdx.x;

  __shared__ float wj[GN_NODES];
  if (t < GN_NODES) {
    const int j = t;
    const float dx = pos[(b * 64 + i) * 2]     - pos[(b * 64 + j) * 2];
    const float dy = pos[(b * 64 + i) * 2 + 1] - pos[(b * 64 + j) * 2 + 1];
    const float d2 = dx * dx + dy * dy;
    const bool adj = ((d2 > 0.f) && (d2 <= 0.0625f)) || (i == j);
    float l = -INFINITY;
    if (adj) {
      const float z = ad_i[(b * 64 + i) * 4 + h] + as_i[(b * 64 + j) * 4 + h];
      l = (z > 0.f) ? z : 0.2f * z;
    }
    wj[t] = l;
  }
  __syncthreads();
  if (t == 0) {   // tiny serial softmax over 64 sources
    float mx = -INFINITY;
    for (int j = 0; j < GN_NODES; ++j) mx = fmaxf(mx, wj[j]);
    float sum = 0.f;
    for (int j = 0; j < GN_NODES; ++j) { const float e = expf(wj[j] - mx); wj[j] = e; sum += e; }
    const float inv = 1.f / sum;
    for (int j = 0; j < GN_NODES; ++j) wj[j] *= inv;
  }
  __syncthreads();

#pragma unroll
  for (int q = 0; q < 4; ++q) {
    const int o = t + 256 * q;
    float acc = 0.f;
    for (int j = 0; j < GN_NODES; ++j)
      acc += wj[j] * xl[(size_t)(b * 64 + j) * FEAT + h * OUTH + o];
    acc += bias[h * OUTH + o];
    const float e = (acc > 0.f) ? acc : (expf(acc) - 1.f);   // ELU
    const size_t off = (size_t)(b * 64 + i) * FEAT + h * OUTH + o;
    yf[off]  = e;
    ybf[off] = (__bf16)e;
  }
}

// ---------------------------------------------------------------------------
// Kernel 6: decoder conv  [1024,16,16,16] -> [1024,8,16,16], ReLU
// ---------------------------------------------------------------------------
__global__ __launch_bounds__(256) void dec_conv_k(
    const float* __restrict__ in, const float* __restrict__ w,
    const float* __restrict__ b, float* __restrict__ out)
{
  const int m  = blockIdx.x;
  const int px = threadIdx.x;
  const int hh = px >> 4, ww = px & 15;

  __shared__ float sx[16][256];
  __shared__ float sw[8 * 16 * 9];
  __shared__ float sb[8];
  for (int i = px; i < 4096; i += 256) sx[i >> 8][i & 255] = in[(size_t)m * 4096 + i];
  for (int i = px; i < 1152; i += 256) sw[i] = w[i];
  if (px < 8) sb[px] = b[px];
  __syncthreads();

  for (int co = 0; co < 8; ++co) {
    float s = sb[co];
#pragma unroll
    for (int ci = 0; ci < 16; ++ci)
#pragma unroll
      for (int kh = -1; kh <= 1; ++kh)
#pragma unroll
        for (int kw = -1; kw <= 1; ++kw) {
          int y = hh + kh, xx = ww + kw;
          if (y >= 0 && y < 16 && xx >= 0 && xx < 16)
            s += sx[ci][y * 16 + xx] * sw[((co * 16 + ci) * 3 + kh + 1) * 3 + kw + 1];
        }
    out[(size_t)m * 2048 + co * 256 + px] = fmaxf(s, 0.f);
  }
}

// ---------------------------------------------------------------------------
// Kernel 7: MLP  [1024,2048] -> relu(@[2048,128]+b1) -> @[128,5]+b2
// ---------------------------------------------------------------------------
__global__ __launch_bounds__(128) void mlp_k(
    const float* __restrict__ hdec, const float* __restrict__ w1,
    const float* __restrict__ b1, const float* __restrict__ w2,
    const float* __restrict__ b2, float* __restrict__ out)
{
  const int r = blockIdx.x, j = threadIdx.x;
  __shared__ float hid[128];
  float s = b1[j];
  const float* row = hdec + (size_t)r * 2048;
  for (int k = 0; k < 2048; ++k) s += row[k] * w1[(size_t)k * 128 + j];
  hid[j] = fmaxf(s, 0.f);
  __syncthreads();
  if (j < 5) {
    float o = b2[j];
    for (int k = 0; k < 128; ++k) o += hid[k] * w2[k * 5 + j];
    out[(size_t)r * 5 + j] = o;
  }
}

// ---------------------------------------------------------------------------
// Host launcher
// ---------------------------------------------------------------------------
extern "C" void kernel_launch(void* const* d_in, const int* in_sizes, int n_in,
                              void* d_out, int out_size, void* d_ws, size_t ws_size,
                              hipStream_t stream) {
  const float* x        = (const float*)d_in[0];
  const float* pos      = (const float*)d_in[1];
  const float* conv1_w  = (const float*)d_in[2];
  const float* conv1_b  = (const float*)d_in[3];
  const float* conv2_w  = (const float*)d_in[4];
  const float* conv2_b  = (const float*)d_in[5];
  const float* gat1_w   = (const float*)d_in[6];
  const float* gat1_as  = (const float*)d_in[7];
  const float* gat1_ad  = (const float*)d_in[8];
  const float* gat1_b   = (const float*)d_in[9];
  const float* gat2_w   = (const float*)d_in[10];
  const float* gat2_as  = (const float*)d_in[11];
  const float* gat2_ad  = (const float*)d_in[12];
  const float* gat2_b   = (const float*)d_in[13];
  const float* dec_w    = (const float*)d_in[14];
  const float* dec_b    = (const float*)d_in[15];
  const float* mlp1_w   = (const float*)d_in[16];
  const float* mlp1_b   = (const float*)d_in[17];
  const float* mlp2_w   = (const float*)d_in[18];
  const float* mlp2_b   = (const float*)d_in[19];
  float* out = (float*)d_out;

  // Workspace layout (~88.1 MB total):
  char* ws = (char*)d_ws;
  float*          h0   = (float*)(ws);                              //  8 MiB
  unsigned short* abf  = (unsigned short*)(ws + (8ull  << 20));     //  8 MiB bf16
  float*          xl   = (float*)(ws + (16ull << 20));              // 16 MiB
  float*          yf   = (float*)(ws + (32ull << 20));              // 16 MiB
  float*          hdec = (float*)(ws + (48ull << 20));              //  8 MiB
  unsigned short* Wt   = (unsigned short*)(ws + (56ull << 20));     // 32 MiB bf16
  float*          as_  = (float*)(ws + (88ull << 20));              // 16 KiB
  float*          ad_  = (float*)(ws + (88ull << 20) + (1 << 14));  // 16 KiB

  // Encoder convs
  conv1_k<<<ROWS, 256, 0, stream>>>(x, conv1_w, conv1_b, h0);
  conv2_k<<<ROWS, 256, 0, stream>>>(h0, conv2_w, conv2_b, abf);

  // GAT layer 1
  wconv_k<<<128 * 128, 256, 0, stream>>>(gat1_w, Wt);
  gemm_bf16_wmma<<<256, 256, 0, stream>>>(abf, Wt, xl);
  attn_coef_k<<<ROWS, 256, 0, stream>>>(xl, gat1_as, gat1_ad, as_, ad_);
  attn_agg_k<<<ROWS * HEADS, 256, 0, stream>>>(xl, pos, as_, ad_, gat1_b, yf, abf);

  // GAT layer 2 (abf now holds layer-1 output in bf16)
  wconv_k<<<128 * 128, 256, 0, stream>>>(gat2_w, Wt);
  gemm_bf16_wmma<<<256, 256, 0, stream>>>(abf, Wt, xl);
  attn_coef_k<<<ROWS, 256, 0, stream>>>(xl, gat2_as, gat2_ad, as_, ad_);
  attn_agg_k<<<ROWS * HEADS, 256, 0, stream>>>(xl, pos, as_, ad_, gat2_b, yf, abf);

  // Decoder + MLP
  dec_conv_k<<<ROWS, 256, 0, stream>>>(yf, dec_w, dec_b, hdec);
  mlp_k<<<ROWS, 128, 0, stream>>>(hdec, mlp1_w, mlp1_b, mlp2_w, mlp2_b, out);
}
